// Decoder_90752658964549
// MI455X (gfx1250) — matrix operand order repaired and checked by reading, not compiled, not run on hardware
//
#include <hip/hip_runtime.h>
#include <hip/hip_bf16.h>
#include <math.h>

typedef __attribute__((ext_vector_type(16))) __bf16 v16bf;
typedef __attribute__((ext_vector_type(8)))  __bf16 v8bf;
typedef __attribute__((ext_vector_type(8)))  float  v8f;

union F8 { v8f v; float f[8]; };

__device__ __forceinline__ v16bf ld_frag(const __bf16* p0, const __bf16* p1) {
  union { v16bf v; v8bf h[2]; } u;
  u.h[0] = *(const v8bf*)p0;
  u.h[1] = *(const v8bf*)p1;
  return u.v;
}

__device__ __forceinline__ v8f wmma_bf16(v16bf a, v16bf b, v8f c) {
  return __builtin_amdgcn_wmma_f32_16x16x32_bf16(false, a, false, b, (short)0, c,
                                                 false, false);
}

// Async memory->LDS copy of 16 bytes per lane (CDNA5 GLOBAL_LOAD_ASYNC_TO_LDS,
// tracked by ASYNCcnt). lds_off is the LDS byte address (low 32 bits of the
// generic pointer, per the CDNA5 aperture rule).
__device__ __forceinline__ void async_load_b128(unsigned lds_off,
                                                const void* gptr) {
  asm volatile("global_load_async_to_lds_b128 %0, %1, off"
               :
               : "v"(lds_off), "v"(gptr)
               : "memory");
}

__device__ __forceinline__ void wait_asynccnt0() {
  asm volatile("s_wait_asynccnt 0" ::: "memory");
}

// ---------------------------------------------------------------- cast f32->bf16
__global__ __launch_bounds__(256) void k_cast(const float* __restrict__ in,
                                              __bf16* __restrict__ out, int n) {
  for (int i = blockIdx.x * blockDim.x + threadIdx.x; i < n;
       i += gridDim.x * blockDim.x)
    out[i] = (__bf16)in[i];
}

// ---------------------------------------------------------------- RMSNorm -> bf16
__global__ __launch_bounds__(256) void k_rmsnorm(const float* __restrict__ x,
                                                 const float* __restrict__ g,
                                                 __bf16* __restrict__ out, int E) {
  __shared__ float red[256];
  const int row = blockIdx.x, tid = threadIdx.x;
  const float* xr = x + (size_t)row * E;
  float4 xv = *(const float4*)(xr + tid * 4);
  float ss = xv.x * xv.x + xv.y * xv.y + xv.z * xv.z + xv.w * xv.w;
  red[tid] = ss;
  __syncthreads();
  for (int s = 128; s > 0; s >>= 1) {
    if (tid < s) red[tid] += red[tid + s];
    __syncthreads();
  }
  float ms = red[0] / (float)E;
  float rs = rsqrtf(ms + 1.1920929e-07f);
  float4 gv = *(const float4*)(g + tid * 4);
  __bf16* orow = out + (size_t)row * E + tid * 4;
  orow[0] = (__bf16)(xv.x * rs * gv.x);
  orow[1] = (__bf16)(xv.y * rs * gv.y);
  orow[2] = (__bf16)(xv.z * rs * gv.z);
  orow[3] = (__bf16)(xv.w * rs * gv.w);
}

// ---------------------------------------------------------------- WMMA GEMM
// C(MxN,f32) = A(MxK,bf16) @ W(KxN,bf16) + bias [+ res]; optional f32/bf16 out.
// Block tile 128x64, 8 waves (4x2), each wave 32x32 (2x2 WMMA tiles), BK=32.
// A tile staged with async global->LDS DMA; B tile transposed via VGPRs.
__global__ __launch_bounds__(256) void k_gemm(const __bf16* __restrict__ A,
                                              const __bf16* __restrict__ W,
                                              const float* __restrict__ bias,
                                              const float* __restrict__ res,
                                              float* __restrict__ outf,
                                              __bf16* __restrict__ outb,
                                              int N, int Kd) {
  __shared__ __bf16 sA[128 * 40];
  __shared__ __bf16 sB[64 * 40];
  const int tid  = threadIdx.x;
  const int lane = tid & 31;
  const int w    = tid >> 5;
  const int wm   = w & 3, wn = w >> 2;
  const int hi   = lane >> 4, l16 = lane & 15;
  const int m0 = blockIdx.y * 128, n0 = blockIdx.x * 64;

  // Per-thread A staging coordinates: 128x32 tile, 512 x 16B chunks.
  const int ar0 = (tid * 8) >> 5,        ac0 = (tid * 8) & 31;
  const int ar1 = ((tid + 256) * 8) >> 5, ac1 = ((tid + 256) * 8) & 31;
  const unsigned ldsA0 = (unsigned)(size_t)(void*)(sA + ar0 * 40 + ac0);
  const unsigned ldsA1 = (unsigned)(size_t)(void*)(sA + ar1 * 40 + ac1);

  F8 acc[2][2];
#pragma unroll
  for (int mi = 0; mi < 2; ++mi)
#pragma unroll
    for (int ni = 0; ni < 2; ++ni)
#pragma unroll
      for (int r = 0; r < 8; ++r) acc[mi][ni].f[r] = 0.f;

  for (int kk = 0; kk < Kd; kk += 32) {
    __syncthreads();
    // stage A tile 128x32 (row-major, padded stride 40) via async DMA
    async_load_b128(ldsA0, A + (size_t)(m0 + ar0) * Kd + kk + ac0);
    async_load_b128(ldsA1, A + (size_t)(m0 + ar1) * Kd + kk + ac1);
    // stage B tile 32x64 transposed -> sB[n][k]
    {
      int idx = tid * 8;
      int kr = idx >> 6, nc = idx & 63;
      v8bf bv = *(const v8bf*)(W + (size_t)(kk + kr) * N + n0 + nc);
#pragma unroll
      for (int e = 0; e < 8; ++e) sB[(nc + e) * 40 + kr] = bv[e];
    }
    if (kk + 32 < Kd)
      __builtin_prefetch(A + (size_t)(m0 + (tid >> 1)) * Kd + kk + 32, 0, 1);
    wait_asynccnt0();
    __syncthreads();

    v16bf af[2], bf[2];
#pragma unroll
    for (int mi = 0; mi < 2; ++mi) {
      const __bf16* ar = sA + (wm * 32 + mi * 16 + l16) * 40 + hi * 8;
      af[mi] = ld_frag(ar, ar + 16);
    }
#pragma unroll
    for (int ni = 0; ni < 2; ++ni) {
      const __bf16* br = sB + (wn * 32 + ni * 16 + l16) * 40 + hi * 8;
      bf[ni] = ld_frag(br, br + 16);
    }
#pragma unroll
    for (int mi = 0; mi < 2; ++mi)
#pragma unroll
      for (int ni = 0; ni < 2; ++ni)
        acc[mi][ni].v = wmma_bf16(af[mi], bf[ni], acc[mi][ni].v);
  }

  // epilogue: C layout -> lanes<16: M=r, lanes>=16: M=8+r; N=l16
#pragma unroll
  for (int mi = 0; mi < 2; ++mi)
#pragma unroll
    for (int ni = 0; ni < 2; ++ni) {
      int col = n0 + wn * 32 + ni * 16 + l16;
      float b = bias[col];
#pragma unroll
      for (int r = 0; r < 8; ++r) {
        int row = m0 + wm * 32 + mi * 16 + r + hi * 8;
        float v = acc[mi][ni].f[r] + b;
        size_t o = (size_t)row * N + col;
        if (res) v += res[o];
        if (outf) outf[o] = v;
        if (outb) outb[o] = (__bf16)v;
      }
    }
}

// ---------------------------------------------------------------- RoPE -> bf16
__global__ __launch_bounds__(256) void k_rope(const float* __restrict__ in,
                                              __bf16* __restrict__ out,
                                              int rows, int nh, int ld, int S) {
  int idx = blockIdx.x * blockDim.x + threadIdx.x;
  int total = rows * nh * 32;
  if (idx >= total) return;
  int j = idx & 31;
  int t = idx >> 5;
  int h = t % nh;
  int row = t / nh;
  int s = row % S;
  float inv = __expf(-(float)j * 0.28782313662425572f);  // ln(1e4)/32
  float ang = (float)s * inv;
  float sn, cs;
  __sincosf(ang, &sn, &cs);
  size_t base = (size_t)row * ld + h * 64;
  float t1 = in[base + j], t2 = in[base + 32 + j];
  out[base + j]      = (__bf16)(t1 * cs - t2 * sn);
  out[base + 32 + j] = (__bf16)(t1 * sn + t2 * cs);
}

// ---------------------------------------------------------------- flash attention
// One wave per 16-row q tile; KV streamed in steps of 32; MQA (1 KV head).
__global__ __launch_bounds__(128) void k_flash(const __bf16* __restrict__ qb,
                                               const __bf16* __restrict__ kb,
                                               const __bf16* __restrict__ vb,
                                               __bf16* __restrict__ ob,
                                               int S, int H) {
  __shared__ __bf16 smem[4 * 3072];  // per wave: V^T 64x40 + P 16x32
  const int lane = threadIdx.x & 31;
  const int w    = threadIdx.x >> 5;
  const int hi   = lane >> 4, l16 = lane & 15;
  const int b = blockIdx.y / H, h = blockIdx.y % H;
  const int i0 = (blockIdx.x * 4 + w) * 16;
  const int ldq = H * 64;

  __bf16* ldsV = smem + w * 3072;         // [d=0..63][k=0..31] stride 40
  __bf16* ldsP = smem + w * 3072 + 2560;  // [m=0..15][k=0..31] stride 32

  const __bf16* Q = qb + ((size_t)b * S) * ldq + h * 64;
  const __bf16* K = kb + (size_t)b * S * 64;
  const __bf16* V = vb + (size_t)b * S * 64;

  const __bf16* qrow = Q + (size_t)(i0 + l16) * ldq + hi * 8;
  v16bf a0 = ld_frag(qrow, qrow + 16);            // HD 0..31
  v16bf a1 = ld_frag(qrow + 32, qrow + 32 + 16);  // HD 32..63

  float mrow[8], lrow[8];
  F8 acc[4];
#pragma unroll
  for (int r = 0; r < 8; ++r) { mrow[r] = -INFINITY; lrow[r] = 0.f; }
#pragma unroll
  for (int c = 0; c < 4; ++c)
#pragma unroll
    for (int r = 0; r < 8; ++r) acc[c].f[r] = 0.f;

  const v8f zero = {0.f, 0.f, 0.f, 0.f, 0.f, 0.f, 0.f, 0.f};
  const int nsteps = (i0 + 15) / 32 + 1;

  for (int stp = 0; stp < nsteps; ++stp) {
    const int jj = stp * 32;
    // stage V[jj..jj+31][0..63] transposed: lane owns key jj+lane
    {
      const __bf16* vr = V + (size_t)(jj + lane) * 64;
#pragma unroll
      for (int d0 = 0; d0 < 64; d0 += 8) {
        v8bf vv = *(const v8bf*)(vr + d0);
#pragma unroll
        for (int e = 0; e < 8; ++e) ldsV[(d0 + e) * 40 + lane] = vv[e];
      }
    }
    // scores: two 16x16 tiles (keys jj.. and jj+16..)
    const __bf16* kr0 = K + (size_t)(jj + l16) * 64 + hi * 8;
    const __bf16* kr1 = K + (size_t)(jj + 16 + l16) * 64 + hi * 8;
    v8f s0 = wmma_bf16(a0, ld_frag(kr0, kr0 + 16), zero);
    s0     = wmma_bf16(a1, ld_frag(kr0 + 32, kr0 + 48), s0);
    v8f s1 = wmma_bf16(a0, ld_frag(kr1, kr1 + 16), zero);
    s1     = wmma_bf16(a1, ld_frag(kr1 + 32, kr1 + 48), s1);

    F8 u0, u1; u0.v = s0; u1.v = s1;
#pragma unroll
    for (int r = 0; r < 8; ++r) {
      int qi = i0 + r + hi * 8;
      int k0 = jj + l16, k1 = jj + 16 + l16;
      float v0 = (k0 <= qi) ? u0.f[r] * 0.125f : -INFINITY;
      float v1 = (k1 <= qi) ? u1.f[r] * 0.125f : -INFINITY;
      float t = fmaxf(v0, v1);
#pragma unroll
      for (int off = 1; off < 16; off <<= 1)
        t = fmaxf(t, __shfl_xor(t, off, 16));
      float nm = fmaxf(mrow[r], t);
      float al = __expf(mrow[r] - nm);
      float p0 = __expf(v0 - nm);
      float p1 = __expf(v1 - nm);
      float ps = p0 + p1;
#pragma unroll
      for (int off = 1; off < 16; off <<= 1) ps += __shfl_xor(ps, off, 16);
      lrow[r] = lrow[r] * al + ps;
      mrow[r] = nm;
#pragma unroll
      for (int c = 0; c < 4; ++c) acc[c].f[r] *= al;
      int m = r + hi * 8;
      ldsP[m * 32 + l16]      = (__bf16)p0;
      ldsP[m * 32 + 16 + l16] = (__bf16)p1;
    }
    asm volatile("s_wait_dscnt 0" ::: "memory");
    __builtin_amdgcn_wave_barrier();

    // P(16x32) @ V(32x64)
    const __bf16* pr = ldsP + l16 * 32 + hi * 8;
    v16bf pa = ld_frag(pr, pr + 16);
#pragma unroll
    for (int c = 0; c < 4; ++c) {
      const __bf16* vr = ldsV + (c * 16 + l16) * 40 + hi * 8;
      acc[c].v = wmma_bf16(pa, ld_frag(vr, vr + 16), acc[c].v);
    }
    __builtin_amdgcn_wave_barrier();
  }

  // normalize + store (B,S,H*HD) bf16
#pragma unroll
  for (int c = 0; c < 4; ++c)
#pragma unroll
    for (int r = 0; r < 8; ++r) {
      int row = i0 + r + hi * 8;
      int col = h * 64 + c * 16 + l16;
      ob[((size_t)b * S + row) * ldq + col] = (__bf16)(acc[c].f[r] / lrow[r]);
    }
}

// ---------------------------------------------------------------- gate: u*gelu(g)
__global__ __launch_bounds__(256) void k_gate(const float* __restrict__ g,
                                              const float* __restrict__ u,
                                              __bf16* __restrict__ out, int n) {
  for (int i = blockIdx.x * blockDim.x + threadIdx.x; i < n;
       i += gridDim.x * blockDim.x) {
    float x = g[i];
    float ge = 0.5f * x * (1.f + erff(x * 0.70710678118654752f));
    out[i] = (__bf16)(u[i] * ge);
  }
}

// ---------------------------------------------------------------- launcher
extern "C" void kernel_launch(void* const* d_in, const int* in_sizes, int n_in,
                              void* d_out, int out_size, void* d_ws, size_t ws_size,
                              hipStream_t stream) {
  (void)in_sizes; (void)n_in; (void)out_size; (void)ws_size;
  const float* x  = (const float*)d_in[0];
  const float* wq = (const float*)d_in[1];  const float* bq = (const float*)d_in[2];
  const float* wk = (const float*)d_in[3];  const float* bk = (const float*)d_in[4];
  const float* wv = (const float*)d_in[5];  const float* bv = (const float*)d_in[6];
  const float* wo = (const float*)d_in[7];  const float* bo = (const float*)d_in[8];
  const float* wg = (const float*)d_in[9];  const float* bg = (const float*)d_in[10];
  const float* wu = (const float*)d_in[11]; const float* bu = (const float*)d_in[12];
  const float* wd = (const float*)d_in[13]; const float* bd = (const float*)d_in[14];
  const float* g1 = (const float*)d_in[15]; const float* g2 = (const float*)d_in[16];

  constexpr int Bn = 2, S = 2048, E = 1024, H = 16, FF = 4096;
  constexpr int M = Bn * S, NQ = H * 64;  // 4096, 1024

  size_t off = 0;
  auto alloc = [&](size_t bytes) -> char* {
    char* p = (char*)d_ws + off;
    off += (bytes + 255) & ~(size_t)255;
    return p;
  };
  __bf16* wqb  = (__bf16*)alloc((size_t)E * NQ * 2);
  __bf16* wkb  = (__bf16*)alloc((size_t)E * 64 * 2);
  __bf16* wvb  = (__bf16*)alloc((size_t)E * 64 * 2);
  __bf16* wob  = (__bf16*)alloc((size_t)NQ * E * 2);
  __bf16* wgb  = (__bf16*)alloc((size_t)E * FF * 2);
  __bf16* wub  = (__bf16*)alloc((size_t)E * FF * 2);
  __bf16* wdb  = (__bf16*)alloc((size_t)FF * E * 2);
  __bf16* hb   = (__bf16*)alloc((size_t)M * E * 2);
  float*  qf   = (float*)alloc((size_t)M * NQ * 4);
  float*  kf   = (float*)alloc((size_t)M * 64 * 4);
  __bf16* qb   = (__bf16*)alloc((size_t)M * NQ * 2);
  __bf16* kbuf = (__bf16*)alloc((size_t)M * 64 * 2);
  __bf16* vbuf = (__bf16*)alloc((size_t)M * 64 * 2);
  __bf16* attb = (__bf16*)alloc((size_t)M * NQ * 2);
  float*  x2   = (float*)alloc((size_t)M * E * 4);
  __bf16* h2b  = (__bf16*)alloc((size_t)M * E * 2);
  float*  gf   = (float*)alloc((size_t)M * FF * 4);
  float*  uf   = (float*)alloc((size_t)M * FF * 4);
  __bf16* mbuf = (__bf16*)alloc((size_t)M * FF * 2);

  auto cast = [&](const float* src, __bf16* dst, int n) {
    int grid = (n + 2047) / 2048;
    k_cast<<<grid, 256, 0, stream>>>(src, dst, n);
  };
  cast(wq, wqb, E * NQ);  cast(wk, wkb, E * 64);  cast(wv, wvb, E * 64);
  cast(wo, wob, NQ * E);  cast(wg, wgb, E * FF);  cast(wu, wub, E * FF);
  cast(wd, wdb, FF * E);

  k_rmsnorm<<<M, 256, 0, stream>>>(x, g1, hb, E);

  k_gemm<<<dim3(NQ / 64, M / 128), 256, 0, stream>>>(hb, wqb, bq, nullptr, qf, nullptr, NQ, E);
  k_gemm<<<dim3(1, M / 128), 256, 0, stream>>>(hb, wkb, bk, nullptr, kf, nullptr, 64, E);
  k_gemm<<<dim3(1, M / 128), 256, 0, stream>>>(hb, wvb, bv, nullptr, nullptr, vbuf, 64, E);

  k_rope<<<(M * H * 32 + 255) / 256, 256, 0, stream>>>(qf, qb, M, H, NQ, S);
  k_rope<<<(M * 32 + 255) / 256, 256, 0, stream>>>(kf, kbuf, M, 1, 64, S);

  k_flash<<<dim3(S / 64, Bn * H), 128, 0, stream>>>(qb, kbuf, vbuf, attb, S, H);

  k_gemm<<<dim3(E / 64, M / 128), 256, 0, stream>>>(attb, wob, bo, x, x2, nullptr, E, NQ);

  k_rmsnorm<<<M, 256, 0, stream>>>(x2, g2, h2b, E);

  k_gemm<<<dim3(FF / 64, M / 128), 256, 0, stream>>>(h2b, wgb, bg, nullptr, gf, nullptr, FF, E);
  k_gemm<<<dim3(FF / 64, M / 128), 256, 0, stream>>>(h2b, wub, bu, nullptr, uf, nullptr, FF, E);

  k_gate<<<8192, 256, 0, stream>>>(gf, uf, mbuf, M * FF);

  k_gemm<<<dim3(E / 64, M / 128), 256, 0, stream>>>(mbuf, wdb, bd, x2, (float*)d_out, nullptr, E, FF);
}
